// PETerNetwork_18949395709982
// MI455X (gfx1250) — matrix operand order repaired and checked by reading, not compiled, not run on hardware
//
#include <hip/hip_runtime.h>
#include <hip/hip_bf16.h>
#include <math.h>

typedef __attribute__((ext_vector_type(16))) _Float16 v16h;
typedef __attribute__((ext_vector_type(8)))  _Float16 v8h;
typedef __attribute__((ext_vector_type(8)))  float    v8f;

#define WMMA_F16(a, b, c) \
  __builtin_amdgcn_wmma_f32_16x16x32_f16(false, (a), false, (b), (short)0, (c), false, false)

// ---------------------------------------------------------------------------
// Fragment helpers. ISA 7.12.2 k-maps are piecewise contiguous:
//   A (16x32 f16): elem e<8 -> k = k0+8g+e ;  e>=8 -> k = k0+16+8g+(e-8)
//   B (32x16 f16): same k-map as A
// With f16 operands each fragment is two 16-byte vector loads
// (global_load_b128 / ds_load_b128), zero conversions in the K-loop.
// ---------------------------------------------------------------------------
__device__ __forceinline__ v16h combine16(v8h lo, v8h hi) {
  v16h a;
#pragma unroll
  for (int i = 0; i < 8; ++i) { a[i] = lo[i]; a[8 + i] = hi[i]; }
  return a;
}

__device__ __forceinline__ v16h frag_a16(const _Float16* row, int k0, int g) {
  v8h lo = *(const v8h*)(row + k0 + 8 * g);
  v8h hi = *(const v8h*)(row + k0 + 16 + 8 * g);
  return combine16(lo, hi);
}
__device__ __forceinline__ v16h frag_b16(const _Float16* row, int k0, int g) {
  v8h lo = *(const v8h*)(row + k0 + 8 * g);
  v8h hi = *(const v8h*)(row + k0 + 16 + 8 * g);
  return combine16(lo, hi);
}

// Scalar A k-map (used only for the gathered edge-conv feature build)
__device__ __forceinline__ int wka(int g, int e) {
  int v = e >> 1, h = e & 1;
  int base = (v < 4) ? (2 * v) : (16 + 2 * (v - 4));
  return base + 8 * g + h;
}
__device__ __forceinline__ int wkb(int g, int e) { return (e < 8) ? (8 * g + e) : (16 + 8 * g + (e - 8)); }

// f32 -> f16 conversion pre-pass
__global__ void cvtf16_kernel(const float* __restrict__ src, _Float16* __restrict__ dst, int n) {
  int i = blockIdx.x * 256 + threadIdx.x;
  if (i < n) dst[i] = (_Float16)src[i];
}

// ---------------------------------------------------------------------------
// Generic WMMA GEMM:  C[M,N] = act(A[M,K] * W[N,K]^T + bias[N])
// A, W pre-converted f16. Requires K % 32 == 0. act: 0=none,1=relu,2=gelu.
// One wave per 16x16 output tile.
// ---------------------------------------------------------------------------
__global__ void gemm_kernel(const _Float16* __restrict__ A, const _Float16* __restrict__ W,
                            const float* __restrict__ bias, float* __restrict__ C,
                            int M, int N, int K, int act) {
  int ntN = (N + 15) >> 4;
  int tm = (int)blockIdx.x / ntN;
  int tn = (int)blockIdx.x % ntN;
  int lane = threadIdx.x & 31;
  int gH = lane >> 4;
  int r = lane & 15;
  int mrow = tm * 16 + r;
  int wrow = tn * 16 + r;
  int am = mrow < M ? mrow : M - 1;
  int wn = wrow < N ? wrow : N - 1;
  const _Float16* arow = A + (size_t)am * K;
  const _Float16* wrowp = W + (size_t)wn * K;

  v8f acc = {0.f, 0.f, 0.f, 0.f, 0.f, 0.f, 0.f, 0.f};
  for (int k0 = 0; k0 < K; k0 += 32) {
    v16h a = frag_a16(arow, k0, gH);
    v16h b = frag_b16(wrowp, k0, gH);
    acc = WMMA_F16(a, b, acc);
  }
  int c = tn * 16 + r;
  if (c >= N) return;
  float bv = bias[c];
#pragma unroll
  for (int i = 0; i < 8; ++i) {
    int m = tm * 16 + 8 * gH + i;
    if (m >= M) continue;
    float y = acc[i] + bv;
    if (act == 1) y = fmaxf(y, 0.f);
    else if (act == 2) y = 0.5f * y * (1.f + erff(y * 0.70710678f));
    C[m * N + c] = y;
  }
}

// ---------------------------------------------------------------------------
// Fused EdgeConv: KNN(K=10) -> feat=[nbr-ctr,ctr] -> GEMM1(64)+BN+leaky (LDS f16)
//                 -> GEMM2(COUT)+BN+leaky -> max over k -> out (cloud,100,COUT)
// One block = 16 points of one cloud. grid = 800*7. K padded 10->16 so each
// point is exactly one WMMA M-tile; pad rows duplicate k=0 (max-neutral).
// Weights pre-converted f16; output type templated (f32 or f16).
// ---------------------------------------------------------------------------
template <int CIN, int COUT, typename OutT>
__global__ void edgeconv_kernel(const float* __restrict__ xin,
    const _Float16* __restrict__ w1, const float* __restrict__ g1, const float* __restrict__ b1,
    const float* __restrict__ m1, const float* __restrict__ v1,
    const _Float16* __restrict__ w2, const float* __restrict__ g2, const float* __restrict__ b2,
    const float* __restrict__ m2, const float* __restrict__ v2,
    OutT* __restrict__ out) {
  constexpr int KF = 2 * CIN;
  __shared__ float    spts[100 * CIN];
  __shared__ int      sidx[16 * 10];
  __shared__ _Float16 sh1[16 * 16 * 64];   // 16 pts x 16 k-slots x 64 ch

  const int cloud = blockIdx.x / 7;
  const int chunk = blockIdx.x % 7;
  const int p0 = chunk * 16;
  const int tid = threadIdx.x;

  for (int i = tid; i < 100 * CIN; i += 256) spts[i] = xin[cloud * 100 * CIN + i];
  __syncthreads();

  // KNN top-10 by dist = 2<xi,xj> - |xi|^2 - |xj|^2 (stable, includes self)
  if (tid < 16) {
    int p = p0 + tid;
    float bestd[10]; int besti[10];
#pragma unroll
    for (int j = 0; j < 10; ++j) { bestd[j] = -3.4e38f; besti[j] = 0; }
    if (p < 100) {
      float np = 0.f;
      for (int c = 0; c < CIN; ++c) { float x = spts[p * CIN + c]; np += x * x; }
      for (int m = 0; m < 100; ++m) {
        float dot = 0.f, nm = 0.f;
        for (int c = 0; c < CIN; ++c) {
          float a = spts[p * CIN + c], q = spts[m * CIN + c];
          dot += a * q; nm += q * q;
        }
        float d = 2.f * dot - np - nm;
        if (d > bestd[9]) {
          int j = 9;
          while (j > 0 && d > bestd[j - 1]) { bestd[j] = bestd[j - 1]; besti[j] = besti[j - 1]; --j; }
          bestd[j] = d; besti[j] = m;
        }
      }
    }
    for (int j = 0; j < 10; ++j) sidx[tid * 10 + j] = besti[j];
  }
  __syncthreads();

  const int wavei = tid >> 5;
  const int lane = tid & 31;
  const int gH = lane >> 4;
  const int r = lane & 15;

  // GEMM1: per point-tile, 4 col-tiles (hidden 64). Feature built on the fly;
  // k0 loop fully unrolled so the CIN-range selects constant-fold.
  for (int t = wavei; t < 16 * 4; t += 8) {
    int pl = t >> 2, ct = t & 3;
    int p = p0 + pl;
    int pe = p < 100 ? p : 99;
    int keff = (r < 10) ? r : 0;           // pad k-slots 10..15 -> k=0
    int nb = sidx[pl * 10 + keff];
    const _Float16* wrowp = w1 + (ct * 16 + r) * KF;
    v8f acc = {0.f, 0.f, 0.f, 0.f, 0.f, 0.f, 0.f, 0.f};
#pragma unroll
    for (int k0 = 0; k0 < KF; k0 += 32) {
      v16h a, b;
#pragma unroll
      for (int e = 0; e < 16; ++e) {
        int ka = k0 + wka(gH, e);
        float av = 0.f;
        if (ka < CIN)      av = spts[nb * CIN + ka] - spts[pe * CIN + ka];
        else if (ka < KF)  av = spts[pe * CIN + (ka - CIN)];
        a[e] = (_Float16)av;
      }
      if constexpr (KF >= 32) {
        b = frag_b16(wrowp, k0, gH);
      } else {
#pragma unroll
        for (int e = 0; e < 16; ++e) {
          int kb = k0 + wkb(gH, e);
          b[e] = (kb < KF) ? wrowp[kb] : (_Float16)0.f;
        }
      }
      acc = WMMA_F16(a, b, acc);
    }
    int c = ct * 16 + r;
    float s  = g1[c] * rsqrtf(v1[c] + 1e-5f);
    float sh = b1[c] - m1[c] * s;
#pragma unroll
    for (int i = 0; i < 8; ++i) {
      float y = acc[i] * s + sh;
      y = (y >= 0.f) ? y : 0.2f * y;
      sh1[(pl * 16 + 8 * gH + i) * 64 + c] = (_Float16)y;
    }
  }
  __syncthreads();

  // GEMM2 + max over k
  constexpr int NCT2 = COUT / 16;
  for (int t = wavei; t < 16 * NCT2; t += 8) {
    int pl = t / NCT2, ct = t % NCT2;
    int p = p0 + pl;
    const _Float16* arow = sh1 + (pl * 16 + r) * 64;
    const _Float16* wrowp = w2 + (ct * 16 + r) * 64;
    v8f acc = {0.f, 0.f, 0.f, 0.f, 0.f, 0.f, 0.f, 0.f};
#pragma unroll
    for (int k0 = 0; k0 < 64; k0 += 32) {
      v16h a = frag_a16(arow, k0, gH);
      v16h b = frag_b16(wrowp, k0, gH);
      acc = WMMA_F16(a, b, acc);
    }
    int c = ct * 16 + r;
    float s  = g2[c] * rsqrtf(v2[c] + 1e-5f);
    float sh = b2[c] - m2[c] * s;
    float mx = -3.4e38f;
#pragma unroll
    for (int i = 0; i < 8; ++i) {
      float y = acc[i] * s + sh;
      y = (y >= 0.f) ? y : 0.2f * y;
      mx = fmaxf(mx, y);
    }
    mx = fmaxf(mx, __shfl_xor(mx, 16, 32));
    if (lane < 16 && p < 100) out[(cloud * 100 + p) * COUT + c] = (OutT)mx;
  }
}

// ---------------------------------------------------------------------------
// Fused shared MLP: relu(bn(x@W1^T+b1)) (LDS f16) -> relu(bn(.@W2^T+b2))
// -> max over 100 points (int-bit atomicMax in LDS; values >= 0 after relu).
// One block per cloud (800 blocks). Rows padded 100->112 (dup row 99).
// Input and weights are f16 (pure b128 fragment loads).
// ---------------------------------------------------------------------------
__global__ void mlp_kernel(const _Float16* __restrict__ hin,
    const _Float16* __restrict__ w1, const float* __restrict__ bias1,
    const float* __restrict__ g1, const float* __restrict__ bb1,
    const float* __restrict__ m1, const float* __restrict__ v1,
    const _Float16* __restrict__ w2, const float* __restrict__ bias2,
    const float* __restrict__ g2, const float* __restrict__ bb2,
    const float* __restrict__ m2, const float* __restrict__ v2,
    float* __restrict__ gout) {
  __shared__ _Float16 sh1[112 * 256];
  __shared__ int smax[256];
  const int cloud = blockIdx.x;
  const int tid = threadIdx.x;
  smax[tid] = 0;                 // == 0.0f bits; relu output >= 0
  const int wavei = tid >> 5, lane = tid & 31, gH = lane >> 4, r = lane & 15;

  // GEMM1: M=112 (7 tiles) x N=256 (16 tiles), K=128
  for (int t = wavei; t < 112; t += 8) {
    int mt = t >> 4, ct = t & 15;
    int row = mt * 16 + r;
    int n = row < 100 ? row : 99;
    const _Float16* arow = hin + (size_t)(cloud * 100 + n) * 128;
    const _Float16* wrowp = w1 + (ct * 16 + r) * 128;
    v8f acc = {0.f, 0.f, 0.f, 0.f, 0.f, 0.f, 0.f, 0.f};
#pragma unroll
    for (int k0 = 0; k0 < 128; k0 += 32) {
      v16h a = frag_a16(arow, k0, gH);
      v16h b = frag_b16(wrowp, k0, gH);
      acc = WMMA_F16(a, b, acc);
    }
    int c = ct * 16 + r;
    float s  = g1[c] * rsqrtf(v1[c] + 1e-5f);
    float sh = bb1[c] - m1[c] * s;
    float bv = bias1[c];
#pragma unroll
    for (int i = 0; i < 8; ++i) {
      float y = (acc[i] + bv) * s + sh;
      y = fmaxf(y, 0.f);
      sh1[(mt * 16 + 8 * gH + i) * 256 + c] = (_Float16)y;
    }
  }
  __syncthreads();

  // GEMM2: K=256, then max over all rows of the cloud
  for (int t = wavei; t < 112; t += 8) {
    int mt = t >> 4, ct = t & 15;
    const _Float16* arow = sh1 + (mt * 16 + r) * 256;
    const _Float16* wrowp = w2 + (ct * 16 + r) * 256;
    v8f acc = {0.f, 0.f, 0.f, 0.f, 0.f, 0.f, 0.f, 0.f};
#pragma unroll
    for (int k0 = 0; k0 < 256; k0 += 32) {
      v16h a = frag_a16(arow, k0, gH);
      v16h b = frag_b16(wrowp, k0, gH);
      acc = WMMA_F16(a, b, acc);
    }
    int c = ct * 16 + r;
    float s  = g2[c] * rsqrtf(v2[c] + 1e-5f);
    float sh = bb2[c] - m2[c] * s;
    float bv = bias2[c];
    float mx = 0.f;
#pragma unroll
    for (int i = 0; i < 8; ++i) {
      float y = (acc[i] + bv) * s + sh;
      y = fmaxf(y, 0.f);
      mx = fmaxf(mx, y);
    }
    mx = fmaxf(mx, __shfl_xor(mx, 16, 32));
    if (lane < 16) atomicMax(&smax[c], __float_as_int(mx));
  }
  __syncthreads();
  gout[cloud * 256 + tid] = __int_as_float(smax[tid]);
}

// ---------------------------------------------------------------------------
// Positional encoding add (in place). grid=800, block=256.
// ---------------------------------------------------------------------------
__global__ void posadd_kernel(float* __restrict__ t) {
  int row = blockIdx.x, c = threadIdx.x;
  int f = row % 25;
  float div = expf((float)(c & ~1) * -0.03597789207803197f);   // -ln(1e4)/256
  float ang = (float)f * div;
  float pe = (c & 1) ? cosf(ang) : sinf(ang);
  t[row * 256 + c] += pe;
}

// ---------------------------------------------------------------------------
// Attention: one wave per (batch, head). S=25, head_dim=32.
// ---------------------------------------------------------------------------
__global__ void attn_kernel(const float* __restrict__ qkv, float* __restrict__ o) {
  __shared__ float sk[25 * 32];
  __shared__ float sv[25 * 32];
  int b = blockIdx.x >> 3;
  int h = blockIdx.x & 7;
  int tid = threadIdx.x;
  for (int i = tid; i < 25 * 32; i += 32) {
    int s = i >> 5, d = i & 31;
    int base = (b * 25 + s) * 768 + h * 32 + d;
    sk[i] = qkv[base + 256];
    sv[i] = qkv[base + 512];
  }
  __syncthreads();
  if (tid < 25) {
    float qv[32];
#pragma unroll
    for (int d = 0; d < 32; ++d) qv[d] = qkv[(b * 25 + tid) * 768 + h * 32 + d];
    float sc[25]; float mx = -3.4e38f;
    for (int j = 0; j < 25; ++j) {
      float dot = 0.f;
#pragma unroll
      for (int d = 0; d < 32; ++d) dot += qv[d] * sk[j * 32 + d];
      sc[j] = dot * 0.1767766953f;   // 1/sqrt(32)
      mx = fmaxf(mx, sc[j]);
    }
    float sum = 0.f;
    for (int j = 0; j < 25; ++j) { float e = expf(sc[j] - mx); sc[j] = e; sum += e; }
    float inv = 1.f / sum;
    float ov[32];
#pragma unroll
    for (int d = 0; d < 32; ++d) ov[d] = 0.f;
    for (int j = 0; j < 25; ++j) {
      float wv = sc[j] * inv;
#pragma unroll
      for (int d = 0; d < 32; ++d) ov[d] += wv * sv[j * 32 + d];
    }
#pragma unroll
    for (int d = 0; d < 32; ++d) o[(b * 25 + tid) * 256 + h * 32 + d] = ov[d];
  }
}

// ---------------------------------------------------------------------------
// out = LayerNorm(x + res) * g + b   (D=256). grid=800, block=256.
// ---------------------------------------------------------------------------
__global__ void addln_kernel(const float* __restrict__ x, const float* __restrict__ res,
                             const float* __restrict__ g, const float* __restrict__ b,
                             float* __restrict__ out) {
  __shared__ float red[256];
  int row = blockIdx.x, c = threadIdx.x;
  float v = x[row * 256 + c] + res[row * 256 + c];
  red[c] = v; __syncthreads();
  for (int s = 128; s > 0; s >>= 1) { if (c < s) red[c] += red[c + s]; __syncthreads(); }
  float mu = red[0] * (1.f / 256.f);
  __syncthreads();
  float d = v - mu;
  red[c] = d * d; __syncthreads();
  for (int s = 128; s > 0; s >>= 1) { if (c < s) red[c] += red[c + s]; __syncthreads(); }
  float var = red[0] * (1.f / 256.f);
  out[row * 256 + c] = d * rsqrtf(var + 1e-5f) * g[c] + b[c];
}

// feat[b,c] = max over 25 frames. grid=32, block=256.
__global__ void maxframes_kernel(const float* __restrict__ t, float* __restrict__ feat) {
  int b = blockIdx.x, c = threadIdx.x;
  float mx = -3.4e38f;
  for (int f = 0; f < 25; ++f) mx = fmaxf(mx, t[(b * 25 + f) * 256 + c]);
  feat[b * 256 + c] = mx;
}

// ---------------------------------------------------------------------------
extern "C" void kernel_launch(void* const* d_in, const int* in_sizes, int n_in,
                              void* d_out, int out_size, void* d_ws, size_t ws_size,
                              hipStream_t stream) {
  auto F = [&](int i) { return (const float*)d_in[i]; };
  const float* x = F(0);
  // ec1: 1..10   ec2: 11..20   mlp: 21..32   tf0: 33..44  tf1: 45..56  cls: 57..62

  // ---- f32 workspace ----
  float* ws    = (float*)d_ws;
  float* hec1  = ws;                    // 800*100*64
  float* tA    = hec1 + 5120000;        // 800*256
  float* tB    = tA + 204800;           // 800*256
  float* qkvb  = tB + 204800;           // 800*768
  float* attno = qkvb + 614400;         // 800*256
  float* proj  = attno + 204800;        // 800*256
  float* ffb   = proj + 204800;         // 800*512
  float* feat  = ffb + 409600;          // 32*256
  float* z1    = feat + 8192;           // 32*256
  float* z2    = z1 + 8192;             // 32*128

  // ---- f16 workspace (16B-aligned: offset is a multiple of 16 bytes) ----
  _Float16* hb = (_Float16*)(ws + 6983680);
  _Float16* hec2h = hb;  hb += 10240000;   // 800*100*128 (EC2 out, MLP in)
  _Float16* actA  = hb;  hb += 409600;     // activation scratch (max 800*512)
  _Float16* wE1a = hb;   hb += 512;
  _Float16* wE1b = hb;   hb += 4096;
  _Float16* wE2a = hb;   hb += 8192;
  _Float16* wE2b = hb;   hb += 8192;
  _Float16* wM1  = hb;   hb += 32768;
  _Float16* wM2  = hb;   hb += 65536;
  _Float16* wQKV[2], *wO[2], *wF1[2], *wF2[2];
  for (int L = 0; L < 2; ++L) {
    wQKV[L] = hb; hb += 196608;
    wO[L]   = hb; hb += 65536;
    wF1[L]  = hb; hb += 131072;
    wF2[L]  = hb; hb += 131072;
  }
  _Float16* wC1 = hb; hb += 65536;
  _Float16* wC2 = hb; hb += 32768;
  _Float16* wC3 = hb; hb += 384;

  auto CVT = [&](const float* s, _Float16* d, int n) {
    cvtf16_kernel<<<(n + 255) / 256, 256, 0, stream>>>(s, d, n);
  };

  // Weight conversion pre-pass (one shot, ~2.5 MB total)
  CVT(F(1),  wE1a, 512);    CVT(F(6),  wE1b, 4096);
  CVT(F(11), wE2a, 8192);   CVT(F(16), wE2b, 8192);
  CVT(F(21), wM1, 32768);   CVT(F(27), wM2, 65536);
  for (int L = 0; L < 2; ++L) {
    int p = 33 + 12 * L;
    CVT(F(p),     wQKV[L], 196608);
    CVT(F(p + 2), wO[L],   65536);
    CVT(F(p + 6), wF1[L],  131072);
    CVT(F(p + 8), wF2[L],  131072);
  }
  CVT(F(57), wC1, 65536);  CVT(F(59), wC2, 32768);  CVT(F(61), wC3, 384);

  edgeconv_kernel<4, 64, float><<<5600, 256, 0, stream>>>(x,
      wE1a, F(2), F(3), F(4), F(5), wE1b, F(7), F(8), F(9), F(10), hec1);
  edgeconv_kernel<64, 128, _Float16><<<5600, 256, 0, stream>>>(hec1,
      wE2a, F(12), F(13), F(14), F(15), wE2b, F(17), F(18), F(19), F(20), hec2h);
  mlp_kernel<<<800, 256, 0, stream>>>(hec2h,
      wM1, F(22), F(23), F(24), F(25), F(26),
      wM2, F(28), F(29), F(30), F(31), F(32), tA);
  posadd_kernel<<<800, 256, 0, stream>>>(tA);

  for (int L = 0; L < 2; ++L) {
    int p = 33 + 12 * L;
    const float *bqkv = F(p + 1), *bo = F(p + 3);
    const float *ln1g = F(p + 4), *ln1b = F(p + 5);
    const float *bf1 = F(p + 7), *bf2 = F(p + 9);
    const float *ln2g = F(p + 10), *ln2b = F(p + 11);

    CVT(tA, actA, 204800);
    gemm_kernel<<<50 * 48, 32, 0, stream>>>(actA, wQKV[L], bqkv, qkvb, 800, 768, 256, 0);
    attn_kernel<<<256, 32, 0, stream>>>(qkvb, attno);
    CVT(attno, actA, 204800);
    gemm_kernel<<<50 * 16, 32, 0, stream>>>(actA, wO[L], bo, proj, 800, 256, 256, 0);
    addln_kernel<<<800, 256, 0, stream>>>(tA, proj, ln1g, ln1b, tB);
    CVT(tB, actA, 204800);
    gemm_kernel<<<50 * 32, 32, 0, stream>>>(actA, wF1[L], bf1, ffb, 800, 512, 256, 2);
    CVT(ffb, actA, 409600);
    gemm_kernel<<<50 * 16, 32, 0, stream>>>(actA, wF2[L], bf2, proj, 800, 256, 512, 0);
    addln_kernel<<<800, 256, 0, stream>>>(tB, proj, ln2g, ln2b, tA);
  }

  maxframes_kernel<<<32, 256, 0, stream>>>(tA, feat);
  CVT(feat, actA, 8192);
  gemm_kernel<<<2 * 16, 32, 0, stream>>>(actA, wC1, F(58), z1, 32, 256, 256, 1);
  CVT(z1, actA, 8192);
  gemm_kernel<<<2 * 8, 32, 0, stream>>>(actA, wC2, F(60), z2, 32, 128, 256, 1);
  CVT(z2, actA, 4096);
  gemm_kernel<<<2 * 1, 32, 0, stream>>>(actA, wC3, F(62), (float*)d_out, 32, 3, 128, 0);
}
